// PointNet2Encoder_48447231098970
// MI455X (gfx1250) — compile-verified, matrix operand order checked by reading, not run on hardware
//
#include <hip/hip_runtime.h>

typedef _Float16 half_t;
typedef _Float16 v16h __attribute__((ext_vector_type(16)));
typedef _Float16 half8 __attribute__((ext_vector_type(8)));
typedef float    v8f  __attribute__((ext_vector_type(8)));

#define BT 32      // B*T point clouds
#define N0 4096    // points per cloud
#define S1 256     // SA1 centroids
#define S2 128     // SA2 centroids
#define NS 32      // neighbors per ball

// ---------------- weight conversion: f32 (cout,cin) -> f16 (cout,kpad) zero-padded ----------------
__global__ void wconv_kernel(const float* __restrict__ w, half_t* __restrict__ wh,
                             int cout, int cin, int kpad) {
  int t = blockIdx.x * blockDim.x + threadIdx.x;
  if (t >= cout * kpad) return;
  int o = t / kpad, k = t % kpad;
  wh[t] = (k < cin) ? (half_t)w[o * cin + k] : (half_t)0.f;
}

// ---------------- farthest point sampling: one block per cloud ----------------
__global__ void fps_kernel(const float* __restrict__ xyz, int N, int S,
                           int* __restrict__ idx, float* __restrict__ nxyz) {
  const int b = blockIdx.x;
  xyz  += (size_t)b * N * 3;
  idx  += (size_t)b * S;
  nxyz += (size_t)b * S * 3;
  const int tid = threadIdx.x;
  const int per = N / 256;           // 16 for N=4096, 1 for N=256
  float mind[16];
  for (int i = 0; i < per; ++i) mind[i] = 1e10f;

  __shared__ float sval[256];
  __shared__ int   sidx[256];
  __shared__ float lastp[3];

  int cur = 0;
  for (int s = 0; s < S; ++s) {
    if (tid == 0) {
      idx[s] = cur;
      float lx = xyz[cur * 3 + 0], ly = xyz[cur * 3 + 1], lz = xyz[cur * 3 + 2];
      lastp[0] = lx; lastp[1] = ly; lastp[2] = lz;
      nxyz[s * 3 + 0] = lx; nxyz[s * 3 + 1] = ly; nxyz[s * 3 + 2] = lz;
    }
    __syncthreads();
    float lx = lastp[0], ly = lastp[1], lz = lastp[2];
    float best = -1.f; int bi = 0;
    for (int i = 0; i < per; ++i) {
      int p = tid + i * 256;
      float dx = xyz[p * 3 + 0] - lx;
      float dy = xyz[p * 3 + 1] - ly;
      float dz = xyz[p * 3 + 2] - lz;
      float d = dx * dx + dy * dy + dz * dz;
      mind[i] = fminf(mind[i], d);
      if (mind[i] > best) { best = mind[i]; bi = p; }
    }
    sval[tid] = best; sidx[tid] = bi;
    __syncthreads();
    for (int off = 128; off > 0; off >>= 1) {
      if (tid < off && sval[tid + off] > sval[tid]) {
        sval[tid] = sval[tid + off]; sidx[tid] = sidx[tid + off];
      }
      __syncthreads();
    }
    cur = sidx[0];
    __syncthreads();
  }
}

// ---------------- ball query: first NS in-radius indices (ascending), pad with first ----------------
__global__ void ballq_kernel(const float* __restrict__ xyz, const float* __restrict__ nxyz,
                             int N, int S, float r2, int* __restrict__ gidx) {
  int t = blockIdx.x * blockDim.x + threadIdx.x;
  if (t >= BT * S) return;
  int b = t / S;
  const float* p = xyz + (size_t)b * N * 3;
  float cx = nxyz[t * 3 + 0], cy = nxyz[t * 3 + 1], cz = nxyz[t * 3 + 2];
  int* out = gidx + (size_t)t * NS;
  int cnt = 0, first = 0; bool found = false;
  for (int j = 0; j < N && cnt < NS; ++j) {
    float dx = p[j * 3 + 0] - cx, dy = p[j * 3 + 1] - cy, dz = p[j * 3 + 2] - cz;
    if (dx * dx + dy * dy + dz * dz < r2) {
      if (!found) { found = true; first = j; }
      out[cnt++] = j;
    }
  }
  for (; cnt < NS; ++cnt) out[cnt] = first;
}

// ---------------- grouping (build zero-padded f16 GEMM inputs) ----------------
__global__ void group1_kernel(const float* __restrict__ xyz, const float* __restrict__ nxyz,
                              const int* __restrict__ gidx, half_t* __restrict__ X) {
  int t = blockIdx.x * blockDim.x + threadIdx.x;    // row = (b*S1+s)*NS + k
  if (t >= BT * S1 * NS) return;
  int sg = t / NS;            // global sample
  int b  = sg / S1;
  int j  = gidx[t];
  const float* p = xyz  + ((size_t)b * N0 + j) * 3;
  const float* c = nxyz + (size_t)sg * 3;
  half_t* row = X + (size_t)t * 32;                 // Kpad = 32
  row[0] = (half_t)(p[0] - c[0]);
  row[1] = (half_t)(p[1] - c[1]);
  row[2] = (half_t)(p[2] - c[2]);
  for (int i = 3; i < 32; ++i) row[i] = (half_t)0.f;
}

__global__ void group2_kernel(const float* __restrict__ xyz, const float* __restrict__ nxyz,
                              const int* __restrict__ gidx, const half_t* __restrict__ feat,
                              half_t* __restrict__ X) {
  int t = blockIdx.x * blockDim.x + threadIdx.x;    // row = (b*S2+s)*NS + k
  if (t >= BT * S2 * NS) return;
  int sg = t / NS;
  int b  = sg / S2;
  int j  = gidx[t];                                 // index into SA1 centroids
  const float*  p = xyz  + ((size_t)b * S1 + j) * 3;
  const float*  c = nxyz + (size_t)sg * 3;
  const half_t* f = feat + ((size_t)b * S1 + j) * 128;
  half_t* row = X + (size_t)t * 160;                // Kpad = 160 (3 + 128 -> 131 padded)
  row[0] = (half_t)(p[0] - c[0]);
  row[1] = (half_t)(p[1] - c[1]);
  row[2] = (half_t)(p[2] - c[2]);
  for (int i = 0; i < 128; ++i) row[3 + i] = f[i];
  for (int i = 131; i < 160; ++i) row[i] = (half_t)0.f;
}

__global__ void group3_kernel(const float* __restrict__ nxyz2, const half_t* __restrict__ feat2,
                              half_t* __restrict__ X) {
  int t = blockIdx.x * blockDim.x + threadIdx.x;    // row = b*S2 + p, total 4096
  if (t >= BT * S2) return;
  half_t* row = X + (size_t)t * 288;                // Kpad = 288 (3 + 256 -> 259 padded)
  row[0] = (half_t)nxyz2[t * 3 + 0];
  row[1] = (half_t)nxyz2[t * 3 + 1];
  row[2] = (half_t)nxyz2[t * 3 + 2];
  const half_t* f = feat2 + (size_t)t * 256;
  for (int i = 0; i < 256; ++i) row[3 + i] = f[i];
  for (int i = 259; i < 288; ++i) row[i] = (half_t)0.f;
}

// ---------------- WMMA fragment load (ISA 7.12.2 16-bit A/B layout) ----------------
// lane<16  -> row/col = lh, halves 0..7 = K k0..k0+7,   8..15 = K k0+16..k0+23
// lane>=16 -> row/col = lh, halves 0..7 = K k0+8..k0+15, 8..15 = K k0+24..k0+31
__device__ __forceinline__ v16h load_frag(const half_t* p) {
  half8 lo = *(const half8*)p;
  half8 hh = *(const half8*)(p + 16);
  v16h r;
  for (int i = 0; i < 8; ++i) { r[i] = lo[i]; r[i + 8] = hh[i]; }
  return r;
}

// ---------------- WMMA GEMM: Y[rows,Cout] = relu(X[rows,K] * W[Cout,K]^T + bias) ----------------
// 256 threads = 8 waves tiled 4(M) x 2(N); wave tile = 32x32 (2x2 fragments, 4 accumulators).
// Block tile = 128 rows x 64 cols. 4 WMMAs per 8 b128 loads -> 2x arithmetic intensity vs 1x1.
__global__ void gemm_wmma_kernel(const half_t* __restrict__ X, const half_t* __restrict__ W,
                                 const float* __restrict__ bias, half_t* __restrict__ Y,
                                 int K, int Cout) {
  const int lane  = threadIdx.x & 31;
  const int wave  = threadIdx.x >> 5;
  const int waveM = wave & 3;
  const int waveN = wave >> 2;
  const int rowBase = blockIdx.x * 128 + waveM * 32;
  const int colBase = blockIdx.y * 64  + waveN * 32;
  const int lh = lane & 15;
  const bool hi = lane >= 16;
  const int koff = hi ? 8 : 0;

  const half_t* a0p = X + (size_t)(rowBase + lh)      * K + koff;
  const half_t* a1p = X + (size_t)(rowBase + 16 + lh) * K + koff;
  const half_t* b0p = W + (size_t)(colBase + lh)      * K + koff;
  const half_t* b1p = W + (size_t)(colBase + 16 + lh) * K + koff;

  v8f acc00 = {0.f,0.f,0.f,0.f,0.f,0.f,0.f,0.f};
  v8f acc01 = acc00, acc10 = acc00, acc11 = acc00;

  for (int k0 = 0; k0 < K; k0 += 32) {
    v16h A0 = load_frag(a0p);
    v16h A1 = load_frag(a1p);
    v16h B0 = load_frag(b0p);
    v16h B1 = load_frag(b1p);
    __builtin_prefetch(a0p + 32, 0, 1);   // global_prefetch_b8 of next K tile
    __builtin_prefetch(a1p + 32, 0, 1);
    __builtin_prefetch(b0p + 32, 0, 1);
    __builtin_prefetch(b1p + 32, 0, 1);
    acc00 = __builtin_amdgcn_wmma_f32_16x16x32_f16(false, A0, false, B0, (short)0, acc00, false, false);
    acc01 = __builtin_amdgcn_wmma_f32_16x16x32_f16(false, A0, false, B1, (short)0, acc01, false, false);
    acc10 = __builtin_amdgcn_wmma_f32_16x16x32_f16(false, A1, false, B0, (short)0, acc10, false, false);
    acc11 = __builtin_amdgcn_wmma_f32_16x16x32_f16(false, A1, false, B1, (short)0, acc11, false, false);
    a0p += 32; a1p += 32; b0p += 32; b1p += 32;
  }

  float bv0 = bias[colBase + lh];
  float bv1 = bias[colBase + 16 + lh];
  for (int r = 0; r < 8; ++r) {
    int m = hi ? 8 + r : r;               // D layout: lanes 0-15 M=r, lanes 16-31 M=8+r
    size_t row0 = (size_t)(rowBase + m) * Cout;
    size_t row1 = (size_t)(rowBase + 16 + m) * Cout;
    Y[row0 + colBase + lh]      = (half_t)fmaxf(acc00[r] + bv0, 0.f);
    Y[row0 + colBase + 16 + lh] = (half_t)fmaxf(acc01[r] + bv1, 0.f);
    Y[row1 + colBase + lh]      = (half_t)fmaxf(acc10[r] + bv0, 0.f);
    Y[row1 + colBase + 16 + lh] = (half_t)fmaxf(acc11[r] + bv1, 0.f);
  }
}

// ---------------- max-pool over K consecutive rows per group ----------------
__global__ void maxpool_kernel(const half_t* __restrict__ Y, int groups, int K, int C,
                               half_t* __restrict__ outH, float* __restrict__ outF) {
  int t = blockIdx.x * blockDim.x + threadIdx.x;
  if (t >= groups * C) return;
  int g = t / C, c = t % C;
  const half_t* p = Y + (size_t)g * K * C + c;
  float m = -1e30f;
  for (int j = 0; j < K; ++j) m = fmaxf(m, (float)p[(size_t)j * C]);
  if (outH) outH[t] = (half_t)m;
  if (outF) outF[t] = m;
}

// ---------------- host orchestration ----------------
extern "C" void kernel_launch(void* const* d_in, const int* in_sizes, int n_in,
                              void* d_out, int out_size, void* d_ws, size_t ws_size,
                              hipStream_t stream) {
  (void)in_sizes; (void)n_in; (void)out_size; (void)ws_size;

  const float* data = (const float*)d_in[0];
  const float* w[9]; const float* bs[9];
  for (int i = 0; i < 9; ++i) { w[i] = (const float*)d_in[1 + 2 * i]; bs[i] = (const float*)d_in[2 + 2 * i]; }

  const int cin [9] = {  3,  64,  64, 131, 128, 128, 259, 256,  512};
  const int cout[9] = { 64,  64, 128, 128, 128, 256, 256, 512, 1024};
  const int kpad[9] = { 32,  64,  64, 160, 128, 128, 288, 256,  512};

  char* ws = (char*)d_ws;
  size_t off = 0;
  auto alloc = [&](size_t bytes) -> char* {
    char* p = ws + off;
    off = (off + bytes + 255) & ~(size_t)255;
    return p;
  };

  half_t* wh[9];
  for (int i = 0; i < 9; ++i) wh[i] = (half_t*)alloc((size_t)cout[i] * kpad[i] * sizeof(half_t));
  int*    idx1  = (int*)   alloc((size_t)BT * S1 * 4);
  float*  nx1   = (float*) alloc((size_t)BT * S1 * 3 * 4);
  int*    gid1  = (int*)   alloc((size_t)BT * S1 * NS * 4);
  int*    idx2  = (int*)   alloc((size_t)BT * S2 * 4);
  float*  nx2   = (float*) alloc((size_t)BT * S2 * 3 * 4);
  int*    gid2  = (int*)   alloc((size_t)BT * S2 * NS * 4);
  half_t* feat1 = (half_t*)alloc((size_t)BT * S1 * 128 * sizeof(half_t));
  half_t* feat2 = (half_t*)alloc((size_t)BT * S2 * 256 * sizeof(half_t));
  const size_t BIG = (size_t)262144 * 128 * sizeof(half_t);   // 64 MiB each
  half_t* A  = (half_t*)alloc(BIG);
  half_t* Bb = (half_t*)alloc(BIG);
  half_t* Cc = (half_t*)alloc(BIG);

  // weight staging
  for (int i = 0; i < 9; ++i) {
    int n = cout[i] * kpad[i];
    wconv_kernel<<<(n + 255) / 256, 256, 0, stream>>>(w[i], wh[i], cout[i], cin[i], kpad[i]);
  }

  // ---- SA1 ----
  fps_kernel<<<BT, 256, 0, stream>>>(data, N0, S1, idx1, nx1);
  ballq_kernel<<<(BT * S1 + 255) / 256, 256, 0, stream>>>(data, nx1, N0, S1, 0.04f, gid1);
  {
    int rows = BT * S1 * NS;                                   // 262144
    group1_kernel<<<(rows + 255) / 256, 256, 0, stream>>>(data, nx1, gid1, A);
    gemm_wmma_kernel<<<dim3(rows / 128,  64 / 64), 256, 0, stream>>>(A,  wh[0], bs[0], Bb, 32, 64);
    gemm_wmma_kernel<<<dim3(rows / 128,  64 / 64), 256, 0, stream>>>(Bb, wh[1], bs[1], Cc, 64, 64);
    gemm_wmma_kernel<<<dim3(rows / 128, 128 / 64), 256, 0, stream>>>(Cc, wh[2], bs[2], A,  64, 128);
    int n = BT * S1 * 128;
    maxpool_kernel<<<(n + 255) / 256, 256, 0, stream>>>(A, BT * S1, NS, 128, feat1, nullptr);
  }

  // ---- SA2 ----
  fps_kernel<<<BT, 256, 0, stream>>>(nx1, S1, S2, idx2, nx2);
  ballq_kernel<<<(BT * S2 + 255) / 256, 256, 0, stream>>>(nx1, nx2, S1, S2, 0.16f, gid2);
  {
    int rows = BT * S2 * NS;                                   // 131072
    group2_kernel<<<(rows + 255) / 256, 256, 0, stream>>>(nx1, nx2, gid2, feat1, Bb);
    gemm_wmma_kernel<<<dim3(rows / 128, 128 / 64), 256, 0, stream>>>(Bb, wh[3], bs[3], Cc, 160, 128);
    gemm_wmma_kernel<<<dim3(rows / 128, 128 / 64), 256, 0, stream>>>(Cc, wh[4], bs[4], A,  128, 128);
    gemm_wmma_kernel<<<dim3(rows / 128, 256 / 64), 256, 0, stream>>>(A,  wh[5], bs[5], Bb, 128, 256);
    int n = BT * S2 * 256;
    maxpool_kernel<<<(n + 255) / 256, 256, 0, stream>>>(Bb, BT * S2, NS, 256, feat2, nullptr);
  }

  // ---- SA3 (group-all) ----
  {
    int rows = BT * S2;                                        // 4096
    group3_kernel<<<(rows + 255) / 256, 256, 0, stream>>>(nx2, feat2, Cc);
    gemm_wmma_kernel<<<dim3(rows / 128,  256 / 64), 256, 0, stream>>>(Cc, wh[6], bs[6], A,  288, 256);
    gemm_wmma_kernel<<<dim3(rows / 128,  512 / 64), 256, 0, stream>>>(A,  wh[7], bs[7], Cc, 256, 512);
    gemm_wmma_kernel<<<dim3(rows / 128, 1024 / 64), 256, 0, stream>>>(Cc, wh[8], bs[8], A,  512, 1024);
    int n = BT * 1024;                                         // 32768 == out_size
    maxpool_kernel<<<(n + 255) / 256, 256, 0, stream>>>(A, BT, S2, 1024, nullptr, (float*)d_out);
  }
}
